// ShapeEncoder_74586402062764
// MI455X (gfx1250) — compile-verified, matrix operand order checked by reading, not run on hardware
//
#include <hip/hip_runtime.h>
#include <cstdint>

typedef __attribute__((ext_vector_type(16))) __bf16 v16bf;
typedef __attribute__((ext_vector_type(8)))  float  v8f;

// ---------------- constants ----------------
namespace cfg {
constexpr int B   = 4;
constexpr int N0  = 8192;
constexpr int C   = 32;
constexpr int R0v = 32;
constexpr int V0  = R0v * R0v * R0v;     // 32768
constexpr int M0  = 1024;
constexpr int KNN = 32;
constexpr int R1v = 16;
constexpr int V1  = R1v * R1v * R1v;     // 4096
constexpr int M1  = 256;
constexpr int N1  = 1024;
constexpr int CH  = 64;                  // padded channel count for grouped features
constexpr int WPS = 32768;               // bf16 elements per weight-pack slot

// workspace layout (float offsets)
constexpr long SZ_COORDS0 = (long)B * 3 * N0;          // 98304
constexpr long SZ_FEAT    = (long)B * C * N0;          // 1048576
constexpr long SZ_NORM    = (long)B * 3 * N0;
constexpr long SZ_CNT     = (long)B * V0;              // 131072
constexpr long SZ_GRID    = (long)B * C * V0;          // 4194304
constexpr long SZ_MISC    = 65536;
constexpr long SZ_CENT0   = (long)B * 3 * M0;          // 12288
constexpr long SZ_CENT1   = 4096;                      // padded
constexpr long SZ_WPACK   = 131072;                    // floats -> 262144 bf16
constexpr long SZ_IDX     = 262144;                    // ints
constexpr long SZ_BIG     = 16777216;                  // 64 MB slab

constexpr long F_COORDS0 = 0;
constexpr long F_FEATSA  = F_COORDS0 + SZ_COORDS0;
constexpr long F_FEATSB  = F_FEATSA + SZ_FEAT;
constexpr long F_NORM    = F_FEATSB + SZ_FEAT;
constexpr long F_CNT     = F_NORM + SZ_NORM;
constexpr long F_GRIDA   = F_CNT + SZ_CNT;
constexpr long F_GRIDB   = F_GRIDA + SZ_GRID;
constexpr long F_VFEAT   = F_GRIDB + SZ_GRID;
constexpr long F_PFEAT   = F_VFEAT + SZ_FEAT;
constexpr long F_MISC    = F_PFEAT + SZ_FEAT;
constexpr long F_CENT0   = F_MISC + SZ_MISC;
constexpr long F_CENT1   = F_CENT0 + SZ_CENT0;
constexpr long F_WPACK   = F_CENT1 + SZ_CENT1;
constexpr long F_IDX     = F_WPACK + SZ_WPACK;
constexpr long F_BIG     = F_IDX + SZ_IDX;
constexpr long F_TOTAL   = F_BIG + SZ_BIG;             // ~30.2M floats ~= 115 MB
} // namespace cfg

// ---------------- device helpers ----------------
__device__ __forceinline__ __bf16 f2bf(float f) {
  unsigned u = __builtin_bit_cast(unsigned, f);
  u += 0x7FFFu + ((u >> 16) & 1u);                      // round-to-nearest-even
  return __builtin_bit_cast(__bf16, (unsigned short)(u >> 16));
}

__device__ __forceinline__ v8f wmma_bf16(v16bf a, v16bf b, v8f c) {
  return __builtin_amdgcn_wmma_f32_16x16x32_bf16(false, a, false, b, (short)0, c, false, false);
}

// K index for A/B fragment element j on half (lane>>4): ISA 7.12.2 16-bit layout
__device__ __forceinline__ int kmap(int j, int half) { return j + (j & 8) + half * 8; }

__device__ __forceinline__ float swishf(float y) { return y / (1.f + __expf(-y)); }

// ---------------- elementwise / utility kernels ----------------
__global__ void k_transpose_in(const float* __restrict__ x, float* __restrict__ coords, int N) {
  int i = blockIdx.x * blockDim.x + threadIdx.x;
  if (i >= cfg::B * N) return;
  int b = i / N, n = i % N;
  for (int d = 0; d < 3; ++d)
    coords[((long)b * 3 + d) * N + n] = x[((long)b * N + n) * 3 + d];
}

__global__ void k_add(const float* __restrict__ a, const float* __restrict__ b,
                      float* __restrict__ y, long total) {
  long i = (long)blockIdx.x * blockDim.x + threadIdx.x;
  if (i < total) y[i] = a[i] + b[i];
}

// ---------------- weight packing ----------------
__global__ void k_pack_pconv_w(const float* __restrict__ w, __bf16* __restrict__ wp,
                               int CO, int CIN, int Kpad) {
  int i = blockIdx.x * blockDim.x + threadIdx.x;
  if (i >= CO * Kpad) return;
  int o = i / Kpad, k = i % Kpad;
  wp[i] = (k < CIN) ? f2bf(w[o * CIN + k]) : f2bf(0.f);
}

// src: [CO][CI][27] -> dst: [CO][27][32] (K zero-padded)
__global__ void k_pack_conv_w(const float* __restrict__ w, __bf16* __restrict__ wp,
                              int CO, int CI) {
  int i = blockIdx.x * blockDim.x + threadIdx.x;
  if (i >= CO * 27 * 32) return;
  int o = i / (27 * 32), t = (i / 32) % 27, c = i % 32;
  wp[i] = (c < CI) ? f2bf(w[((long)o * CI + c) * 27 + t]) : f2bf(0.f);
}

// ---------------- WMMA GEMM kernels ----------------
// Y[b,o,n] = sum_k Wp[o,k] * X[b,k,n] + bias[o] (+ res)
// One wave computes a 32x16 output tile: two weight fragments share one
// activation fragment (halves f32->bf16 convert cost on the activation side).
__global__ void k_pconv_wmma(const __bf16* __restrict__ Wp, const float* __restrict__ X,
                             const float* __restrict__ bias, const float* res,
                             float* __restrict__ Y, int CO, int CIN, int Kpad, int NS) {
  int lane = threadIdx.x & 31, half = lane >> 4, lm = lane & 15;
  int nt = blockIdx.x, ot = blockIdx.y, b = blockIdx.z;
  int n = nt * 16 + lm;
  int row0 = ot * 32 + lm, row1 = row0 + 16;
  v8f acc0 = {0, 0, 0, 0, 0, 0, 0, 0};
  v8f acc1 = {0, 0, 0, 0, 0, 0, 0, 0};
  for (int kb = 0; kb < Kpad; kb += 32) {
    v16bf a0, a1, bb;
#pragma unroll
    for (int j = 0; j < 16; ++j) {
      int k = kb + kmap(j, half);
      a0[j] = Wp[(long)row0 * Kpad + k];
      a1[j] = Wp[(long)row1 * Kpad + k];
      float xv = (k < CIN) ? X[((long)b * CIN + k) * NS + n] : 0.f;
      bb[j] = f2bf(xv);
    }
    acc0 = wmma_bf16(a0, bb, acc0);
    acc1 = wmma_bf16(a1, bb, acc1);
  }
#pragma unroll
  for (int g = 0; g < 8; ++g) {
    int m0 = ot * 32 + g + half * 8;
    int m1 = m0 + 16;
    long o0 = ((long)b * CO + m0) * NS + nt * 16 + lm;
    long o1 = ((long)b * CO + m1) * NS + nt * 16 + lm;
    float v0 = acc0[g] + bias[m0];
    float v1 = acc1[g] + bias[m1];
    if (res) { v0 += res[o0]; v1 += res[o1]; }
    Y[o0] = v0;
    Y[o1] = v1;
  }
}

// implicit-GEMM 3x3x3 conv over r^3 grid, 32 in-ch (padded), 32 out-ch.
// One wave computes all 32 output channels for 16 voxels: per tap, one
// activation fragment feeds two weight fragments (2 WMMAs).
// Wc: [32][27][32] bf16, X/Y: [B][32][V]
__global__ void k_conv3d_wmma(const __bf16* __restrict__ Wc, const float* __restrict__ X,
                              const float* __restrict__ bias, float* __restrict__ Y, int r) {
  int lane = threadIdx.x & 31, half = lane >> 4, lm = lane & 15;
  int wt = r >> 4;
  int t = blockIdx.x;
  int w0 = (t % wt) * 16;
  int rem = t / wt;
  int h = rem % r, d = rem / r;
  int b = blockIdx.y;
  int V = r * r * r;
  int row0 = lm, row1 = lm + 16;
  v8f acc0 = {0, 0, 0, 0, 0, 0, 0, 0};
  v8f acc1 = {0, 0, 0, 0, 0, 0, 0, 0};
  for (int tap = 0; tap < 27; ++tap) {
    int dd = d + tap / 9 - 1;
    int hh = h + (tap / 3) % 3 - 1;
    if (dd < 0 || dd >= r || hh < 0 || hh >= r) continue;  // uniform across wave
    int ww = w0 + lm + (tap % 3) - 1;
    bool inb = (ww >= 0 && ww < r);
    long base = (long)b * 32 * V + ((long)(dd * r + hh) * r + ww);
    v16bf a0, a1, bb;
#pragma unroll
    for (int j = 0; j < 16; ++j) {
      int k = kmap(j, half);
      a0[j] = Wc[((long)row0 * 27 + tap) * 32 + k];
      a1[j] = Wc[((long)row1 * 27 + tap) * 32 + k];
      float xv = inb ? X[base + (long)k * V] : 0.f;
      bb[j] = f2bf(xv);
    }
    acc0 = wmma_bf16(a0, bb, acc0);
    acc1 = wmma_bf16(a1, bb, acc1);
  }
  int col = (d * r + h) * r + w0;
#pragma unroll
  for (int g = 0; g < 8; ++g) {
    int m0 = g + half * 8;
    int m1 = m0 + 16;
    Y[((long)b * 32 + m0) * V + col + lm] = acc0[g] + bias[m0];
    Y[((long)b * 32 + m1) * V + col + lm] = acc1[g] + bias[m1];
  }
}

// S[i,j] = sum_c q[c,i] * k[c,j]. One wave: one q fragment vs two key tiles.
__global__ void k_attn_qk(const float* __restrict__ q, const float* __restrict__ kf,
                          float* __restrict__ S, int NS) {
  int lane = threadIdx.x & 31, half = lane >> 4, lm = lane & 15;
  int it = blockIdx.x, jt = blockIdx.y;  // jt covers 32 keys
  v16bf a, b0, b1;
#pragma unroll
  for (int j = 0; j < 16; ++j) {
    int c = kmap(j, half);
    a[j]  = f2bf(q[(long)c * NS + it * 16 + lm]);
    b0[j] = f2bf(kf[(long)c * NS + jt * 32 + lm]);
    b1[j] = f2bf(kf[(long)c * NS + jt * 32 + 16 + lm]);
  }
  v8f acc0 = {0, 0, 0, 0, 0, 0, 0, 0};
  v8f acc1 = {0, 0, 0, 0, 0, 0, 0, 0};
  acc0 = wmma_bf16(a, b0, acc0);
  acc1 = wmma_bf16(a, b1, acc1);
#pragma unroll
  for (int g = 0; g < 8; ++g) {
    long rowo = ((long)it * 16 + g + half * 8) * NS + jt * 32;
    S[rowo + lm] = acc0[g];
    S[rowo + 16 + lm] = acc1[g];
  }
}

// h[c,i] = sum_m v[c,m] * S[i,m]. One wave: all 32 channels for 16 queries;
// per 32-key step one S fragment feeds two v fragments (2 WMMAs).
__global__ void k_attn_av(const float* __restrict__ vf, const float* __restrict__ S,
                          float* __restrict__ h, int NS) {
  int lane = threadIdx.x & 31, half = lane >> 4, lm = lane & 15;
  int it = blockIdx.x;
  v8f acc0 = {0, 0, 0, 0, 0, 0, 0, 0};
  v8f acc1 = {0, 0, 0, 0, 0, 0, 0, 0};
  const float* srow = S + ((long)it * 16 + lm) * NS;
  for (int mb = 0; mb < NS; mb += 32) {
    __builtin_prefetch((const void*)(srow + mb + 32), 0, 1);  // global_prefetch_b8
    v16bf a0, a1, bb;
#pragma unroll
    for (int j = 0; j < 16; ++j) {
      int m = mb + kmap(j, half);
      a0[j] = f2bf(vf[(long)lm * NS + m]);
      a1[j] = f2bf(vf[(long)(lm + 16) * NS + m]);
      bb[j] = f2bf(srow[m]);
    }
    acc0 = wmma_bf16(a0, bb, acc0);
    acc1 = wmma_bf16(a1, bb, acc1);
  }
#pragma unroll
  for (int g = 0; g < 8; ++g) {
    int m0 = g + half * 8;
    h[(long)m0 * NS + it * 16 + lm] = acc0[g];
    h[(long)(m0 + 16) * NS + it * 16 + lm] = acc1[g];
  }
}

// ---------------- voxelization ----------------
__global__ void k_vox_stats(const float* __restrict__ coords, int N, float* __restrict__ out) {
  __shared__ float sh[256];
  __shared__ float mean[3];
  int b = blockIdx.x, t = threadIdx.x;
  for (int d = 0; d < 3; ++d) {
    float s = 0.f;
    for (int i = t; i < N; i += 256) s += coords[((long)b * 3 + d) * N + i];
    sh[t] = s; __syncthreads();
    for (int k = 128; k > 0; k >>= 1) { if (t < k) sh[t] += sh[t + k]; __syncthreads(); }
    if (t == 0) mean[d] = sh[0] / (float)N;
    __syncthreads();
  }
  float mx = 0.f;
  for (int i = t; i < N; i += 256) {
    float dx = coords[((long)b * 3 + 0) * N + i] - mean[0];
    float dy = coords[((long)b * 3 + 1) * N + i] - mean[1];
    float dz = coords[((long)b * 3 + 2) * N + i] - mean[2];
    mx = fmaxf(mx, dx * dx + dy * dy + dz * dz);
  }
  sh[t] = mx; __syncthreads();
  for (int k = 128; k > 0; k >>= 1) { if (t < k) sh[t] = fmaxf(sh[t], sh[t + k]); __syncthreads(); }
  if (t == 0) {
    out[b * 4 + 0] = mean[0]; out[b * 4 + 1] = mean[1]; out[b * 4 + 2] = mean[2];
    out[b * 4 + 3] = sqrtf(sh[0]) * 2.f;
  }
}

__global__ void k_vox_scatter(const float* __restrict__ coords, const float* __restrict__ feats,
                              int CI, int N, int r, int V, const float* __restrict__ vs,
                              float* grid, float* cnt, float* __restrict__ norm) {
  int i = blockIdx.x * blockDim.x + threadIdx.x;
  if (i >= cfg::B * N) return;
  int b = i / N, n = i % N;
  float sc = vs[b * 4 + 3];
  int vc[3];
  for (int d = 0; d < 3; ++d) {
    float v = (coords[((long)b * 3 + d) * N + n] - vs[b * 4 + d]) / sc + 0.5f;
    v *= (float)r;
    v = fminf(fmaxf(v, 0.f), (float)(r - 1));
    norm[((long)b * 3 + d) * N + n] = v;
    vc[d] = (int)rintf(v);
  }
  int flat = (vc[0] * r + vc[1]) * r + vc[2];
  atomicAdd(&cnt[(long)b * V + flat], 1.f);
  for (int c = 0; c < CI; ++c)
    atomicAdd(&grid[((long)b * 32 + c) * V + flat], feats[((long)b * CI + c) * N + n]);
}

__global__ void k_vox_div(float* grid, const float* __restrict__ cnt, int V) {
  int i = blockIdx.x * blockDim.x + threadIdx.x;
  if (i >= cfg::B * V) return;
  int b = i / V, v = i % V;
  float inv = 1.f / fmaxf(cnt[(long)b * V + v], 1.f);
  for (int c = 0; c < 32; ++c) grid[((long)b * 32 + c) * V + v] *= inv;
}

__global__ void k_devox(const float* __restrict__ grid, const float* __restrict__ norm,
                        float* __restrict__ out, int r, int N) {
  int i = blockIdx.x * blockDim.x + threadIdx.x;
  if (i >= cfg::B * N) return;
  int b = i / N, n = i % N;
  float nc[3], f[3];
  int c0[3], c1[3];
  for (int d = 0; d < 3; ++d) {
    nc[d] = norm[((long)b * 3 + d) * N + n];
    int a = (int)floorf(nc[d]);
    a = min(max(a, 0), r - 1);
    c0[d] = a; c1[d] = min(a + 1, r - 1);
    f[d] = nc[d] - (float)a;
  }
  float acc[32];
  for (int c = 0; c < 32; ++c) acc[c] = 0.f;
  for (int dx = 0; dx < 2; ++dx)
    for (int dy = 0; dy < 2; ++dy)
      for (int dz = 0; dz < 2; ++dz) {
        int ix = dx ? c1[0] : c0[0];
        int iy = dy ? c1[1] : c0[1];
        int iz = dz ? c1[2] : c0[2];
        float w = (dx ? f[0] : 1.f - f[0]) * (dy ? f[1] : 1.f - f[1]) * (dz ? f[2] : 1.f - f[2]);
        long idx = (long)(ix * r + iy) * r + iz;
        long base = (long)b * 32 * r * r * r + idx;
        long V = (long)r * r * r;
        for (int c = 0; c < 32; ++c) acc[c] += grid[base + (long)c * V] * w;
      }
  for (int c = 0; c < 32; ++c) out[((long)b * 32 + c) * N + n] = acc[c];
}

// ---------------- group norm ----------------
__global__ void k_gn_stats(const float* __restrict__ X, float* __restrict__ stats, int C, int V) {
  __shared__ float sh[256];
  int bg = blockIdx.x, b = bg >> 3, g = bg & 7;
  int Cg = C / 8;
  long L = (long)Cg * V;
  const float* p = X + ((long)b * C + (long)g * Cg) * V;
  float a = 0.f, q = 0.f;
  for (long i = threadIdx.x; i < L; i += 256) { float x = p[i]; a += x; q += x * x; }
  sh[threadIdx.x] = a; __syncthreads();
  for (int s = 128; s > 0; s >>= 1) { if (threadIdx.x < s) sh[threadIdx.x] += sh[threadIdx.x + s]; __syncthreads(); }
  float suma = sh[0]; __syncthreads();
  sh[threadIdx.x] = q; __syncthreads();
  for (int s = 128; s > 0; s >>= 1) { if (threadIdx.x < s) sh[threadIdx.x] += sh[threadIdx.x + s]; __syncthreads(); }
  if (threadIdx.x == 0) {
    float mean = suma / (float)L;
    float var = sh[0] / (float)L - mean * mean;
    stats[bg * 2 + 0] = mean;
    stats[bg * 2 + 1] = rsqrtf(var + 1e-5f);
  }
}

__global__ void k_gn_apply(const float* X, float* Y, const float* __restrict__ gamma,
                           const float* __restrict__ beta, const float* __restrict__ stats,
                           int C, int V, int doswish, long total) {
  long i = (long)blockIdx.x * blockDim.x + threadIdx.x;
  if (i >= total) return;
  int c = (int)((i / V) % C);
  int b = (int)(i / ((long)C * V));
  int g = c / (C / 8);
  float y = (X[i] - stats[(b * 8 + g) * 2]) * stats[(b * 8 + g) * 2 + 1] * gamma[c] + beta[c];
  if (doswish) y = swishf(y);
  Y[i] = y;
}

// ---------------- SE ----------------
__global__ void k_se_mean(const float* __restrict__ X, int V, float* __restrict__ s) {
  __shared__ float sh[256];
  int bc = blockIdx.x;
  const float* p = X + (long)bc * V;
  float a = 0.f;
  for (int i = threadIdx.x; i < V; i += 256) a += p[i];
  sh[threadIdx.x] = a; __syncthreads();
  for (int k = 128; k > 0; k >>= 1) { if (threadIdx.x < k) sh[threadIdx.x] += sh[threadIdx.x + k]; __syncthreads(); }
  if (threadIdx.x == 0) s[bc] = sh[0] / (float)V;
}

__global__ void k_se_mlp(const float* __restrict__ se1, const float* __restrict__ se2,
                         const float* __restrict__ s, float* __restrict__ sev) {
  __shared__ float r1[4];
  int b = blockIdx.x, t = threadIdx.x;
  if (t < 4) {
    float a = 0.f;
    for (int c = 0; c < 32; ++c) a += se1[t * 32 + c] * s[b * 32 + c];
    r1[t] = fmaxf(a, 0.f);
  }
  __syncthreads();
  if (t < 32) {
    float a = 0.f;
    for (int j = 0; j < 4; ++j) a += se2[t * 4 + j] * r1[j];
    sev[b * 32 + t] = 1.f / (1.f + __expf(-a));
  }
}

__global__ void k_se_scale(float* X, const float* __restrict__ sev, int V, long total) {
  long i = (long)blockIdx.x * blockDim.x + threadIdx.x;
  if (i >= total) return;
  int bc = (int)(i / V);
  X[i] *= sev[bc];
}

// ---------------- FPS / ball query / grouping ----------------
__global__ void k_fps(const float* __restrict__ coords, int N, int M, int* __restrict__ outIdx) {
  __shared__ float dists[8192];
  __shared__ float bv[256];
  __shared__ int bi[256];
  __shared__ float lastp[3];
  __shared__ int lastidx;
  int b = blockIdx.x, t = threadIdx.x;
  for (int i = t; i < N; i += 256) dists[i] = 1e10f;
  if (t == 0) { outIdx[b * M] = 0; lastidx = 0; }
  __syncthreads();
  for (int m = 1; m < M; ++m) {
    if (t < 3) lastp[t] = coords[((long)b * 3 + t) * N + lastidx];
    __syncthreads();
    float bestv = -1.f; int besti = 0;
    for (int i = t; i < N; i += 256) {
      float dx = coords[((long)b * 3 + 0) * N + i] - lastp[0];
      float dy = coords[((long)b * 3 + 1) * N + i] - lastp[1];
      float dz = coords[((long)b * 3 + 2) * N + i] - lastp[2];
      float d = fminf(dists[i], dx * dx + dy * dy + dz * dz);
      dists[i] = d;
      if (d > bestv) { bestv = d; besti = i; }
    }
    bv[t] = bestv; bi[t] = besti;
    __syncthreads();
    for (int s = 128; s > 0; s >>= 1) {
      if (t < s && bv[t + s] > bv[t]) { bv[t] = bv[t + s]; bi[t] = bi[t + s]; }
      __syncthreads();
    }
    if (t == 0) { outIdx[b * M + m] = bi[0]; lastidx = bi[0]; }
    __syncthreads();
  }
}

__global__ void k_gather_centers(const float* __restrict__ coords, const int* __restrict__ cidx,
                                 float* __restrict__ centers, int N, int M) {
  int i = blockIdx.x * blockDim.x + threadIdx.x;
  if (i >= cfg::B * M) return;
  int b = i / M, m = i % M;
  int p = cidx[b * M + m];
  for (int d = 0; d < 3; ++d)
    centers[((long)b * 3 + d) * M + m] = coords[((long)b * 3 + d) * N + p];
}

__global__ void k_ball_query(const float* __restrict__ centers, const float* __restrict__ coords,
                             int M, int N, float r2, int* __restrict__ out) {
  int i = blockIdx.x * blockDim.x + threadIdx.x;
  if (i >= cfg::B * M) return;
  int b = i / M, m = i % M;
  float cx = centers[((long)b * 3 + 0) * M + m];
  float cy = centers[((long)b * 3 + 1) * M + m];
  float cz = centers[((long)b * 3 + 2) * M + m];
  int* o = out + ((long)b * M + m) * cfg::KNN;
  int cnt = 0;
  for (int n = 0; n < N && cnt < cfg::KNN; ++n) {
    float dx = coords[((long)b * 3 + 0) * N + n] - cx;
    float dy = coords[((long)b * 3 + 1) * N + n] - cy;
    float dz = coords[((long)b * 3 + 2) * N + n] - cz;
    if (dx * dx + dy * dy + dz * dz < r2) o[cnt++] = n;
  }
  int fill = (cnt > 0) ? o[0] : 0;
  for (int j = cnt; j < cfg::KNN; ++j) o[j] = fill;
}

// G: [B][64][M*K], channels 0..2 = rel coords, 3..3+CIN-1 = gathered features
__global__ void k_group_concat(const float* __restrict__ coords, const float* __restrict__ centers,
                               const float* __restrict__ feats, const int* __restrict__ idx,
                               float* __restrict__ G, int N, int M, int CIN) {
  int i = blockIdx.x * blockDim.x + threadIdx.x;
  int total = cfg::B * M * cfg::KNN;
  if (i >= total) return;
  int k = i % cfg::KNN;
  int m = (i / cfg::KNN) % M;
  int b = i / (M * cfg::KNN);
  int p = idx[((long)b * M + m) * cfg::KNN + k];
  long S = (long)M * cfg::KNN;
  long col = (long)m * cfg::KNN + k;
  for (int d = 0; d < 3; ++d)
    G[((long)b * cfg::CH + d) * S + col] =
        coords[((long)b * 3 + d) * N + p] - centers[((long)b * 3 + d) * M + m];
  for (int c = 0; c < CIN; ++c)
    G[((long)b * cfg::CH + 3 + c) * S + col] = feats[((long)b * CIN + c) * N + p];
}

__global__ void k_max_lastk(const float* __restrict__ g, float* __restrict__ out,
                            int C, int M, int K, int total) {
  int i = blockIdx.x * blockDim.x + threadIdx.x;
  if (i >= total) return;
  int m = i % M, c = (i / M) % C, b = i / (M * C);
  const float* p = g + ((long)(b * C + c) * M + m) * K;
  float mx = p[0];
  for (int k = 1; k < K; ++k) mx = fmaxf(mx, p[k]);
  out[((long)b * C + c) * M + m] = mx;
}

__global__ void k_max_lastm(const float* __restrict__ f, float* __restrict__ out, int C, int M) {
  int i = blockIdx.x * blockDim.x + threadIdx.x;
  if (i >= cfg::B * C) return;
  const float* p = f + (long)i * M;
  float mx = p[0];
  for (int m = 1; m < M; ++m) mx = fmaxf(mx, p[m]);
  out[i] = mx;
}

// ---------------- softmax / head ----------------
__global__ void k_softmax_rows(float* S, int NS) {
  __shared__ float red[256];
  int t = threadIdx.x;
  float* r = S + (long)blockIdx.x * NS;
  float mx = -1e30f;
  for (int i = t; i < NS; i += 256) mx = fmaxf(mx, r[i]);
  red[t] = mx; __syncthreads();
  for (int s = 128; s > 0; s >>= 1) { if (t < s) red[t] = fmaxf(red[t], red[t + s]); __syncthreads(); }
  mx = red[0]; __syncthreads();
  float sum = 0.f;
  for (int i = t; i < NS; i += 256) { float e = __expf(r[i] - mx); r[i] = e; sum += e; }
  red[t] = sum; __syncthreads();
  for (int s = 128; s > 0; s >>= 1) { if (t < s) red[t] += red[t + s]; __syncthreads(); }
  float inv = 1.f / red[0];
  __syncthreads();
  for (int i = t; i < NS; i += 256) r[i] *= inv;
}

__global__ void k_head(const float* __restrict__ W, const float* __restrict__ bias,
                       const float* __restrict__ f, float* __restrict__ out) {
  int b = blockIdx.x, o = threadIdx.x;  // 256 threads
  float a = bias[o];
  for (int c = 0; c < 128; ++c) a += W[o * 128 + c] * f[b * 128 + c];
  out[b * 256 + o] = a;
}

// ---------------- host orchestration ----------------
namespace {

struct WS {
  float *coords0, *featsA, *featsB, *norm, *cnt, *gridA, *gridB, *vfeat, *pfeat;
  float *misc, *cent0, *cent1, *big;
  __bf16* wp;
  int *cidx0, *bq0, *cidx1, *bq1;
};

inline int cdiv(long a, int b) { return (int)((a + b - 1) / b); }

void run_gn(const float* x, float* y, const float* g, const float* bt, float* stats,
            int C, int V, int sw, hipStream_t s) {
  k_gn_stats<<<dim3(cfg::B * 8), 256, 0, s>>>(x, stats, C, V);
  long total = (long)cfg::B * C * V;
  k_gn_apply<<<dim3(cdiv(total, 256)), 256, 0, s>>>(x, y, g, bt, stats, C, V, sw, total);
}

void run_pconv(const float* w, const float* bias, const float* X, const float* res, float* Y,
               int CO, int CIN, int Kpad, int NS, __bf16* slot, hipStream_t s) {
  k_pack_pconv_w<<<dim3(cdiv((long)CO * Kpad, 256)), 256, 0, s>>>(w, slot, CO, CIN, Kpad);
  k_pconv_wmma<<<dim3(NS / 16, CO / 32, cfg::B), 32, 0, s>>>(slot, X, bias, res, Y, CO, CIN,
                                                             Kpad, NS);
}

// params P: c1w c1b g1g g1b c2w c2b g2g g2b se1 se2 pw pb pgg pgb [qw qb kw kb vw vb ow ob agg agb]
void do_pvconv(const float* const* P, const float* inFeats, int CI, const float* coords,
               int Npts, int r, bool att, WS& w, float* outFeats, hipStream_t s) {
  int V = r * r * r;
  float* stats = w.misc;        // 64
  float* vstat = w.misc + 256;  // 16
  float* seS = w.misc + 512;    // 128
  float* seV = w.misc + 768;    // 128
  // ---- voxelize ----
  hipMemsetAsync(w.gridA, 0, (size_t)cfg::B * 32 * V * sizeof(float), s);
  hipMemsetAsync(w.cnt, 0, (size_t)cfg::B * V * sizeof(float), s);
  k_vox_stats<<<dim3(cfg::B), 256, 0, s>>>(coords, Npts, vstat);
  k_vox_scatter<<<dim3(cdiv((long)cfg::B * Npts, 256)), 256, 0, s>>>(
      coords, inFeats, CI, Npts, r, V, vstat, w.gridA, w.cnt, w.norm);
  k_vox_div<<<dim3(cdiv((long)cfg::B * V, 256)), 256, 0, s>>>(w.gridA, w.cnt, V);
  // ---- conv1 + gn + swish ----
  k_pack_conv_w<<<dim3(cdiv(32 * 27 * 32, 256)), 256, 0, s>>>(P[0], w.wp, 32, CI);
  k_conv3d_wmma<<<dim3(V / 16, cfg::B), 32, 0, s>>>(w.wp, w.gridA, P[1], w.gridB, r);
  run_gn(w.gridB, w.gridB, P[2], P[3], stats, 32, V, 1, s);
  // ---- conv2 + gn ----
  k_pack_conv_w<<<dim3(cdiv(32 * 27 * 32, 256)), 256, 0, s>>>(P[4], w.wp + cfg::WPS, 32, 32);
  k_conv3d_wmma<<<dim3(V / 16, cfg::B), 32, 0, s>>>(w.wp + cfg::WPS, w.gridB, P[5], w.gridA, r);
  run_gn(w.gridA, w.gridA, P[6], P[7], stats, 32, V, att ? 0 : 1, s);
  float* h = w.gridA;
  if (att) {
    // q,k,v,h slabs inside gridB
    long BQ = (long)cfg::B * 32 * V;
    float *q = w.gridB, *kf = w.gridB + BQ, *vf = w.gridB + 2 * BQ, *hb = w.gridB + 3 * BQ;
    run_pconv(P[14], P[15], w.gridA, nullptr, q, 32, 32, 32, V, w.wp + 2 * cfg::WPS, s);
    run_pconv(P[16], P[17], w.gridA, nullptr, kf, 32, 32, 32, V, w.wp + 3 * cfg::WPS, s);
    run_pconv(P[18], P[19], w.gridA, nullptr, vf, 32, 32, 32, V, w.wp + 4 * cfg::WPS, s);
    for (int b = 0; b < cfg::B; ++b) {
      long off = (long)b * 32 * V;
      k_attn_qk<<<dim3(V / 16, V / 32), 32, 0, s>>>(q + off, kf + off, w.big, V);
      k_softmax_rows<<<dim3(V), 256, 0, s>>>(w.big, V);
      k_attn_av<<<dim3(V / 16), 32, 0, s>>>(vf + off, w.big, hb + off, V);
    }
    // o-proj + residual (xf = gridA), then gn+swish -> gridB slab 0
    run_pconv(P[20], P[21], hb, w.gridA, q, 32, 32, 32, V, w.wp + 5 * cfg::WPS, s);
    run_gn(q, q, P[22], P[23], stats, 32, V, 1, s);
    h = q;
  }
  // ---- SE ----
  k_se_mean<<<dim3(cfg::B * 32), 256, 0, s>>>(h, V, seS);
  k_se_mlp<<<dim3(cfg::B), 32, 0, s>>>(P[8], P[9], seS, seV);
  long totV = (long)cfg::B * 32 * V;
  k_se_scale<<<dim3(cdiv(totV, 256)), 256, 0, s>>>(h, seV, V, totV);
  // ---- devox ----
  k_devox<<<dim3(cdiv((long)cfg::B * Npts, 128)), 128, 0, s>>>(h, w.norm, w.vfeat, r, Npts);
  // ---- point branch ----
  run_pconv(P[10], P[11], inFeats, nullptr, w.pfeat, 32, CI, 32, Npts, w.wp + 6 * cfg::WPS, s);
  run_gn(w.pfeat, w.pfeat, P[12], P[13], stats, 32, Npts, 1, s);
  long totP = (long)cfg::B * 32 * Npts;
  k_add<<<dim3(cdiv(totP, 256)), 256, 0, s>>>(w.vfeat, w.pfeat, outFeats, totP);
}

// SA: P = [w0 b0 gg0 gb0 w1 b1 gg1 gb1], layer0: (32, 3+CIN), layer1: (CO2, 32)
void do_sa(const float* const* P, int CO2, const float* feats, int CIN, const float* coords,
           int Npts, int M, float radius, WS& w, int* cidx, int* bq, float* centers,
           float* outFeats, hipStream_t s) {
  int NS = M * cfg::KNN;
  float* stats = w.misc;
  k_fps<<<dim3(cfg::B), 256, 0, s>>>(coords, Npts, M, cidx);
  k_gather_centers<<<dim3(cdiv((long)cfg::B * M, 128)), 128, 0, s>>>(coords, cidx, centers, Npts, M);
  k_ball_query<<<dim3(cdiv((long)cfg::B * M, 128)), 128, 0, s>>>(centers, coords, M, Npts,
                                                                 radius * radius, bq);
  hipMemsetAsync(w.big, 0, (size_t)cfg::B * cfg::CH * NS * sizeof(float), s);
  k_group_concat<<<dim3(cdiv((long)cfg::B * M * cfg::KNN, 256)), 256, 0, s>>>(
      coords, centers, feats, bq, w.big, Npts, M, CIN);
  // layer 0: 64(pad 3+CIN) -> 32
  run_pconv(P[0], P[1], w.big, nullptr, w.gridA, 32, cfg::CH, cfg::CH, NS, w.wp + 7 * cfg::WPS, s);
  run_gn(w.gridA, w.gridA, P[2], P[3], stats, 32, NS, 1, s);
  // layer 1: 32 -> CO2
  run_pconv(P[4], P[5], w.gridA, nullptr, w.gridB, CO2, 32, 32, NS, w.wp + 7 * cfg::WPS, s);
  run_gn(w.gridB, w.gridB, P[6], P[7], stats, CO2, NS, 1, s);
  int tot = cfg::B * CO2 * M;
  k_max_lastk<<<dim3(cdiv(tot, 256)), 256, 0, s>>>(w.gridB, outFeats, CO2, M, cfg::KNN, tot);
}

} // namespace

extern "C" void kernel_launch(void* const* d_in, const int* in_sizes, int n_in,
                              void* d_out, int out_size, void* d_ws, size_t ws_size,
                              hipStream_t stream) {
  using namespace cfg;
  if ((long)ws_size < F_TOTAL * 4 || n_in < 71) return;

  float* W = (float*)d_ws;
  WS w;
  w.coords0 = W + F_COORDS0;
  w.featsA  = W + F_FEATSA;
  w.featsB  = W + F_FEATSB;
  w.norm    = W + F_NORM;
  w.cnt     = W + F_CNT;
  w.gridA   = W + F_GRIDA;
  w.gridB   = W + F_GRIDB;
  w.vfeat   = W + F_VFEAT;
  w.pfeat   = W + F_PFEAT;
  w.misc    = W + F_MISC;
  w.cent0   = W + F_CENT0;
  w.cent1   = W + F_CENT1;
  w.big     = W + F_BIG;
  w.wp      = (__bf16*)(W + F_WPACK);
  int* ireg = (int*)(W + F_IDX);
  w.cidx0 = ireg;
  w.bq0   = ireg + 4096;
  w.cidx1 = ireg + 4096 + B * M0 * KNN;
  w.bq1   = w.cidx1 + 1024;

  const float* x = (const float*)d_in[0];
  const float* const* P0 = (const float* const*)&d_in[1];   // l0_pv0 (14)
  const float* const* P1 = (const float* const*)&d_in[15];  // l0_pv1 (14)
  const float* const* S0 = (const float* const*)&d_in[29];  // l0_sa  (8)
  const float* const* P2 = (const float* const*)&d_in[37];  // l1_pv  (24, att)
  const float* const* S1 = (const float* const*)&d_in[61];  // l1_sa  (8)
  const float* headW = (const float*)d_in[69];
  const float* headB = (const float*)d_in[70];
  float* out = (float*)d_out;

  // coords (B,3,N0)
  k_transpose_in<<<dim3(cdiv((long)B * N0, 256)), 256, 0, stream>>>(x, w.coords0, N0);

  // l0_pv0: feats=coords (CI=3) -> featsA
  do_pvconv(P0, w.coords0, 3, w.coords0, N0, R0v, false, w, w.featsA, stream);
  // l0_pv1: featsA -> featsB
  do_pvconv(P1, w.featsA, 32, w.coords0, N0, R0v, false, w, w.featsB, stream);
  // l0_sa: (featsB, coords0) -> featsA (B,32,M0), centers0
  do_sa(S0, 32, w.featsB, 32, w.coords0, N0, M0, 0.1f, w, w.cidx0, w.bq0, w.cent0,
        w.featsA, stream);
  // l1_pv (attention): featsA @ centers0 -> featsB (B,32,N1)
  do_pvconv(P2, w.featsA, 32, w.cent0, N1, R1v, true, w, w.featsB, stream);
  // l1_sa: (featsB, centers0) -> vfeat (B,128,M1)
  do_sa(S1, 128, w.featsB, 32, w.cent0, N1, M1, 0.2f, w, w.cidx1, w.bq1, w.cent1,
        w.vfeat, stream);
  // global max over M1 -> f (B,128), then head
  float* fvec = w.misc + 1024;
  k_max_lastm<<<dim3(cdiv(B * 128, 128)), 128, 0, stream>>>(w.vfeat, fvec, 128, M1);
  k_head<<<dim3(B), 256, 0, stream>>>(headW, headB, fvec, out);
}
